// FNetBlock_12051678232667
// MI455X (gfx1250) — matrix operand order repaired and checked by reading, not compiled, not run on hardware
//
#include <hip/hip_runtime.h>

typedef __attribute__((ext_vector_type(16))) _Float16 v16h;
typedef __attribute__((ext_vector_type(8)))  _Float16 v8h;
typedef __attribute__((ext_vector_type(8)))  float    v8f;

#define LAG     32
#define Lsz     512
#define Dsz     64
#define NR      257      // rfft output length along k-axis
#define NPAD    512      // sort width (padded with +inf)
#define ROWS    16       // (b,l) rows per workgroup
#define THREADS 256      // 8 wave32s

// raw v_cos_f32: input in revolutions, cos(2*pi*x)
__device__ __forceinline__ float cos_rev(float x) {
    return __builtin_amdgcn_cosf(x);
}

__device__ __forceinline__ v8f wmma_f16(v16h a, v16h b, v8f c) {
    return __builtin_amdgcn_wmma_f32_16x16x32_f16(false, a, false, b,
                                                  (short)0, c, false, false);
}

__global__ __launch_bounds__(THREADS)
void fnet_fused_kernel(const float* __restrict__ x,
                       const float* __restrict__ mask,
                       const float* __restrict__ W,
                       const float* __restrict__ bptr,
                       float* __restrict__ out)
{
    __shared__ __align__(16) _Float16 sMask[ROWS][Lsz]; // 16 KB, f16 A-operand tile
    __shared__ float sM[ROWS][NPAD];                    // 32 KB, M rows / sort buffer
    __shared__ float sX[ROWS][Dsz];                     // 4 KB
    __shared__ float sW[2 * LAG];

    const int t    = threadIdx.x;
    const int wave = t >> 5;
    const int lane = t & 31;
    const int R    = blockIdx.x * ROWS;                 // global row base over (b*L + l)

    // ---- stage 0: stage mask tile (f32 -> f16), x tile, W into LDS ----
    for (int i = t; i < ROWS * Lsz; i += THREADS) {
        int m = i >> 9, k = i & 511;
        sMask[m][k] = (_Float16)mask[(size_t)(R + m) * Lsz + k];
    }
    for (int i = t; i < ROWS * Dsz; i += THREADS) {
        int m = i >> 6, d = i & 63;
        sX[m][d] = x[(size_t)(R + m) * Dsz + d];
    }
    if (t < 2 * LAG) sW[t] = W[t];
    __syncthreads();

    // ---- stage 1: M[m][n] = sum_k mask[m][k]*cos(2*pi*n*k/512) via WMMA f16 ----
    // A (16-bit 16x32): lane m=l&15, kgrp=l>>4; halves = two aligned b128 LDS chunks.
    // B: lanes 0-15 K=k0..k0+15, lanes 16-31 K=k0+16..k0+31, column n0+(l&15).
    // Cosines: float phase accumulator in revolutions, one v_cos + one v_add each.
    // Junk columns 257..271 get overwritten by the +inf pad before sorting.
    const int mrow = lane & 15;
    const int kgrp = lane >> 4;
    {
        const int ncolA = wave * 16 + (lane & 15);        // tiles 0..7
        const int ncolB = (wave + 8) * 16 + (lane & 15);  // tiles 8..15
        const float stepA = (float)ncolA * (1.0f / 512.0f);
        const float stepB = (float)ncolB * (1.0f / 512.0f);
        v8f cA = {}, cB = {};
        for (int k0 = 0; k0 < Lsz; k0 += 32) {
            v8h lo = *(const v8h*)&sMask[mrow][k0 + kgrp * 16];
            v8h hi = *(const v8h*)&sMask[mrow][k0 + kgrp * 16 + 8];
            v16h a = __builtin_shufflevector(lo, hi,
                        0,1,2,3,4,5,6,7,8,9,10,11,12,13,14,15);

            const int kb = k0 + kgrp * 16;
            float pA = (float)((kb * ncolA) & 511) * (1.0f / 512.0f);
            float pB = (float)((kb * ncolB) & 511) * (1.0f / 512.0f);
            v16h bmA, bmB;
            #pragma unroll
            for (int h = 0; h < 16; ++h) {
                bmA[h] = (_Float16)cos_rev(pA);
                bmB[h] = (_Float16)cos_rev(pB);
                pA += stepA;
                pB += stepB;
            }
            cA = wmma_f16(a, bmA, cA);
            cB = wmma_f16(a, bmB, cB);
        }
        #pragma unroll
        for (int i = 0; i < 8; ++i) {
            sM[i + 8 * kgrp][ncolA] = cA[i];
            sM[i + 8 * kgrp][ncolB] = cB[i];
        }
    }
    // leftover N-tile 16 (columns 256..271) — wave 0 only (uniform; execz-branched)
    if (wave == 0) {
        const int ncol = 256 + (lane & 15);
        const float step = (float)ncol * (1.0f / 512.0f);
        v8f c = {};
        for (int k0 = 0; k0 < Lsz; k0 += 32) {
            v8h lo = *(const v8h*)&sMask[mrow][k0 + kgrp * 16];
            v8h hi = *(const v8h*)&sMask[mrow][k0 + kgrp * 16 + 8];
            v16h a = __builtin_shufflevector(lo, hi,
                        0,1,2,3,4,5,6,7,8,9,10,11,12,13,14,15);
            float p = (float)(((k0 + kgrp * 16) * ncol) & 511) * (1.0f / 512.0f);
            v16h bm;
            #pragma unroll
            for (int h = 0; h < 16; ++h) {
                bm[h] = (_Float16)cos_rev(p);
                p += step;
            }
            c = wmma_f16(a, bm, c);
        }
        #pragma unroll
        for (int i = 0; i < 8; ++i)
            sM[i + 8 * kgrp][ncol] = c[i];
    }
    __syncthreads();   // M rows complete (each row's columns came from all waves)

    // ================= per-wave tail: wave w owns rows 2w, 2w+1 =================
    // Wave32 LDS ops are in-order within a wave -> no further workgroup barriers.
    const int r0 = 2 * wave;
    const int r1 = r0 + 1;

    // pad columns 257..511 with +inf (also kills junk cols 257..271)
    for (int n = NR + lane; n < NPAD; n += 32) {
        sM[r0][n] = __builtin_inff();
        sM[r1][n] = __builtin_inff();
    }

    // Xc kept in registers: lane's 4 values map 1:1 to its 4 output elements
    float xc[4];
    #pragma unroll
    for (int p = 0; p < 4; ++p) {
        int o  = lane + p * 32;          // 0..127
        int rr = r0 + (o >> 6);
        int f  = o & 63;
        float step = (float)f * (1.0f / 64.0f);
        float ph = 0.f, acc = 0.f;
        #pragma unroll 8
        for (int d = 0; d < Dsz; ++d) {
            acc = __builtin_fmaf(sX[rr][d], cos_rev(ph), acc);
            ph += step;
        }
        xc[p] = acc;
    }

    // barrier-free ascending bitonic sort of this wave's 2 rows (512 each)
    for (int k = 2; k <= NPAD; k <<= 1) {
        for (int j = k >> 1; j > 0; j >>= 1) {
            #pragma unroll 1
            for (int p = 0; p < 16; ++p) {         // 2 rows * 256 pairs / 32 lanes
                int idx = lane + p * 32;           // 0..511
                int rr  = r0 + (idx >> 8);
                int q   = idx & 255;
                int i   = ((q & ~(j - 1)) << 1) | (q & (j - 1));
                int l2  = i | j;
                float a  = sM[rr][i];
                float b2 = sM[rr][l2];
                bool up  = ((i & k) == 0);
                if ((a > b2) == up) { sM[rr][i] = b2; sM[rr][l2] = a; }
            }
        }
    }

    // Spos/Sneg: LAG==32 terms -> one term per lane, xor-shuffle reduction
    float sp0 = sW[lane] * sM[r0][lane]       + sW[32 + lane] * sM[r0][225 + lane];
    float sn0 = sW[lane] * sM[r0][256 - lane] + sW[32 + lane] * sM[r0][31 - lane];
    float sp1 = sW[lane] * sM[r1][lane]       + sW[32 + lane] * sM[r1][225 + lane];
    float sn1 = sW[lane] * sM[r1][256 - lane] + sW[32 + lane] * sM[r1][31 - lane];
    #pragma unroll
    for (int off = 16; off > 0; off >>= 1) {
        sp0 += __shfl_xor(sp0, off, 32);
        sn0 += __shfl_xor(sn0, off, 32);
        sp1 += __shfl_xor(sp1, off, 32);
        sn1 += __shfl_xor(sn1, off, 32);
    }

    // emb = Xc * (Xc >= 0 ? Spos : Sneg) + b
    const float b0 = bptr[0];
    #pragma unroll
    for (int p = 0; p < 4; ++p) {
        int o  = lane + p * 32;
        int rr = r0 + (o >> 6);
        int f  = o & 63;
        float v = xc[p];
        float s = (o >> 6) ? (v >= 0.f ? sp1 : sn1)
                           : (v >= 0.f ? sp0 : sn0);
        out[(size_t)(R + rr) * Dsz + f] = __builtin_fmaf(v, s, b0);
    }
}

extern "C" void kernel_launch(void* const* d_in, const int* in_sizes, int n_in,
                              void* d_out, int out_size, void* d_ws, size_t ws_size,
                              hipStream_t stream) {
    const float* x    = (const float*)d_in[0];
    const float* mask = (const float*)d_in[1];
    const float* W    = (const float*)d_in[2];
    const float* b    = (const float*)d_in[3];
    float* out = (float*)d_out;

    dim3 grid((4 * Lsz) / ROWS);   // 128 workgroups, 16 (b,l) rows each
    fnet_fused_kernel<<<grid, THREADS, 0, stream>>>(x, mask, W, b, out);
}
